// GroupedQueryAttention_43224550867135
// MI455X (gfx1250) — matrix operand order repaired and checked by reading, not compiled, not run on hardware
//
#include <hip/hip_runtime.h>

typedef __bf16 bf16;
typedef __attribute__((ext_vector_type(16))) __bf16 v16bf;
typedef __attribute__((ext_vector_type(8)))  __bf16 v8bf;
typedef __attribute__((ext_vector_type(8)))  float  v8f;

#define DEV static __device__ __forceinline__

// ---- optional CDNA5 async global->LDS path (fallback: sync copy) ----------
#if defined(__HIP_DEVICE_COMPILE__) && defined(__gfx1250__) && \
    __has_builtin(__builtin_amdgcn_global_load_async_to_lds_b128) && \
    __has_builtin(__builtin_amdgcn_s_wait_asynccnt)
#define HAVE_ASYNC_LDS 1
#else
#define HAVE_ASYNC_LDS 0
#endif

// builtin takes pointers to GCC-style int4 vectors in AS1 (global) / AS3 (LDS)
typedef int gv4i __attribute__((vector_size(16)));
typedef __attribute__((address_space(1))) gv4i gv4i_g;
typedef __attribute__((address_space(3))) gv4i gv4i_l;

DEV void copy16_to_lds(bf16* lds, const bf16* g) {
#if HAVE_ASYNC_LDS
  // flat LDS address keeps the LDS byte offset in its low 32 bits
  __builtin_amdgcn_global_load_async_to_lds_b128(
      (gv4i_g*)(uintptr_t)g, (gv4i_l*)(unsigned)(uintptr_t)lds, 0, 0);
#else
  *(v8bf*)lds = *(const v8bf*)g;
#endif
}
DEV void async_wait0() {
#if HAVE_ASYNC_LDS
  __builtin_amdgcn_s_wait_asynccnt(0);
#endif
}

DEV bf16 f2bf(float f) {
  unsigned u = __builtin_bit_cast(unsigned, f);
  unsigned r = u + 0x7FFFu + ((u >> 16) & 1u);   // round-to-nearest-even
  unsigned short h = (unsigned short)(r >> 16);
  return __builtin_bit_cast(bf16, h);
}
DEV float bf2f(bf16 b) {
  unsigned short h = __builtin_bit_cast(unsigned short, b);
  unsigned u = ((unsigned)h) << 16;
  return __builtin_bit_cast(float, u);
}
DEV v8f v8f_zero() {
  v8f z = {0.f, 0.f, 0.f, 0.f, 0.f, 0.f, 0.f, 0.f};
  return z;
}

union V16 { v16bf v; v8bf h[2]; };

// A-operand (16x32 bf16): lanes 0-15 rows M with K {0..7,16..23},
// lanes 16-31 same rows with K {8..15,24..31}.  (ISA 7.12.2)
DEV v16bf load_frag_a(const bf16* base, int stride, int row0, int k0, int lane) {
  const int r = lane & 15, hs = lane >> 4;
  V16 f;
  f.h[0] = *(const v8bf*)(base + (row0 + r) * stride + k0 + hs * 8);
  f.h[1] = *(const v8bf*)(base + (row0 + r) * stride + k0 + 16 + hs * 8);
  return f.v;
}
// B-operand (32x16 bf16) from a row-major [N x K] tile:
// lane holds column n = lane&15; lanes 0-15 K {0..7,16..23}, lanes 16-31 K {8..15,24..31}.
DEV v16bf load_frag_b(const bf16* base, int stride, int col0, int k0, int lane) {
  const int r = lane & 15, hs = lane >> 4;
  V16 f;
  f.h[0] = *(const v8bf*)(base + (col0 + r) * stride + k0 + hs * 8);
  f.h[1] = *(const v8bf*)(base + (col0 + r) * stride + k0 + 16 + hs * 8);
  return f.v;
}
DEV v8f wmma_bf16(v16bf a, v16bf b, v8f c) {
  return __builtin_amdgcn_wmma_f32_16x16x32_bf16(false, a, false, b, (short)0, c,
                                                 false, false);
}

// ---------------------------------------------------------------- converters
__global__ __launch_bounds__(256)
void cvt_f32_bf16(const float* __restrict__ in, bf16* __restrict__ out, int n8) {
  int i = blockIdx.x * 256 + threadIdx.x;
  if (i >= n8) return;
  const float4* p = (const float4*)in + (size_t)i * 2;
  float4 a = p[0], b = p[1];
  union { v8bf v; bf16 e[8]; } o;
  o.e[0] = f2bf(a.x); o.e[1] = f2bf(a.y); o.e[2] = f2bf(a.z); o.e[3] = f2bf(a.w);
  o.e[4] = f2bf(b.x); o.e[5] = f2bf(b.y); o.e[6] = f2bf(b.z); o.e[7] = f2bf(b.w);
  *((v8bf*)out + i) = o.v;
}

// tiled transpose-convert: in K x Ncols row-major (fp32) -> out Ncols x K (bf16)
// coalesced on both sides via a 32x33 LDS tile.
__global__ __launch_bounds__(256)
void tcvt_f32_bf16(const float* __restrict__ in, bf16* __restrict__ out,
                   int K, int Ncols) {
  __shared__ float tile[32][33];
  const int k0 = blockIdx.x * 32;   // K dimension
  const int n0 = blockIdx.y * 32;   // N dimension
  const int lx = threadIdx.x & 31;
  const int ly = threadIdx.x >> 5;  // 8 rows per pass
#pragma unroll
  for (int yy = 0; yy < 32; yy += 8)
    tile[ly + yy][lx] = in[(size_t)(k0 + ly + yy) * Ncols + n0 + lx];
  __syncthreads();
#pragma unroll
  for (int yy = 0; yy < 32; yy += 8)
    out[(size_t)(n0 + ly + yy) * K + k0 + lx] = f2bf(tile[lx][ly + yy]);
}

// ---------------------------------------------------------------- GEMM
// C[M,N] = A[M,K] * Bt[N,K]^T  (bf16 in, f32 accum, bf16 or f32 out)
// 128x128 block tile, K-step 32, double-buffered async LDS staging.
template <bool F32OUT>
__global__ __launch_bounds__(256)
void gemm_bt(const bf16* __restrict__ A, const bf16* __restrict__ Bt,
             void* __restrict__ Cout, int M, int N, int K) {
  __shared__ alignas(16) bf16 As[2][128 * 40];
  __shared__ alignas(16) bf16 Bs[2][128 * 40];

  const int tid = threadIdx.x;
  const int lane = tid & 31;
  const int wave = tid >> 5;
  const int wrow = wave >> 1;   // 0..3 : 32-row strip
  const int wcol = wave & 1;    // 0..1 : 64-col strip
  const int n0 = blockIdx.x * 128;
  const int m0 = blockIdx.y * 128;

  v8f acc[2][4];
#pragma unroll
  for (int i = 0; i < 2; ++i)
#pragma unroll
    for (int j = 0; j < 4; ++j) acc[i][j] = v8f_zero();

  auto stage = [&](int k0, int buf) {
#pragma unroll
    for (int g = 0; g < 2; ++g) {
      int gid = g * 256 + tid;
      int row = gid >> 2;          // 4 groups of 8 per 32-wide row
      int c = (gid & 3) * 8;
      copy16_to_lds(&As[buf][row * 40 + c],
                    A + (size_t)(m0 + row) * K + k0 + c);
      copy16_to_lds(&Bs[buf][row * 40 + c],
                    Bt + (size_t)(n0 + row) * K + k0 + c);
    }
  };

  stage(0, 0);
  async_wait0();
  __syncthreads();

  const int nsteps = K >> 5;
  for (int t = 0; t < nsteps; ++t) {
    const int buf = t & 1;
    if (t + 1 < nsteps) stage((t + 1) * 32, buf ^ 1);
    if (t + 2 < nsteps) {  // speculative L2 prefetch two slabs ahead
      int prow = tid & 127;
      __builtin_prefetch(A + (size_t)(m0 + prow) * K + (t + 2) * 32, 0, 1);
      __builtin_prefetch(Bt + (size_t)(n0 + prow) * K + (t + 2) * 32, 0, 1);
    }

    v16bf a0 = load_frag_a(As[buf], 40, wrow * 32 + 0, 0, lane);
    v16bf a1 = load_frag_a(As[buf], 40, wrow * 32 + 16, 0, lane);
#pragma unroll
    for (int nt = 0; nt < 4; ++nt) {
      v16bf b = load_frag_b(Bs[buf], 40, wcol * 64 + nt * 16, 0, lane);
      acc[0][nt] = wmma_bf16(a0, b, acc[0][nt]);
      acc[1][nt] = wmma_bf16(a1, b, acc[1][nt]);
    }
    async_wait0();
    __syncthreads();
  }

  const int r = lane & 15, hs = lane >> 4;
#pragma unroll
  for (int mt = 0; mt < 2; ++mt)
#pragma unroll
    for (int nt = 0; nt < 4; ++nt)
#pragma unroll
      for (int i = 0; i < 8; ++i) {
        int row = m0 + wrow * 32 + mt * 16 + hs * 8 + i;
        int col = n0 + wcol * 64 + nt * 16 + r;
        if (F32OUT)
          ((float*)Cout)[(size_t)row * N + col] = acc[mt][nt][i];
        else
          ((bf16*)Cout)[(size_t)row * N + col] = f2bf(acc[mt][nt][i]);
      }
}

// ---------------------------------------------------------------- RoPE
// Reference uses cos for both terms: out[j]=(x1-x2)*c ; out[j+64]=(x1+x2)*c
__global__ __launch_bounds__(256)
void rope_inplace(bf16* __restrict__ t, int heads) {
  int idx = blockIdx.x * 256 + threadIdx.x;
  int j = idx & 63;
  int th = idx >> 6;
  int hh = th % heads;
  int tok = th / heads;
  if (tok >= 4096) return;
  int pos = tok & 2047;  // position within sequence (N = 2048)
  float inv_freq = __expf(-(float)(2 * j) * (9.2103403719761836f / 128.0f));
  float c = __cosf((float)pos * inv_freq);
  bf16* p = t + ((size_t)tok * heads + hh) * 128;
  float x1 = bf2f(p[j]);
  float x2 = bf2f(p[j + 64]);
  p[j] = f2bf((x1 - x2) * c);
  p[j + 64] = f2bf((x1 + x2) * c);
}

// ---------------------------------------------------------------- attention
// One block: 128 query rows for one (batch, head). 8 waves, 16 q-rows each.
// KV tiles of 32 rows, online softmax, O accumulated over 128 cols per wave.
__global__ __launch_bounds__(256)
void attn_fwd(const bf16* __restrict__ Q, const bf16* __restrict__ Kb,
              const bf16* __restrict__ Vb, bf16* __restrict__ O) {
  __shared__ alignas(16) bf16 Qs[128 * 136];  // 128 q-rows x HD
  __shared__ alignas(16) bf16 Ks[32 * 136];   // 32 kv-rows x HD
  __shared__ alignas(16) bf16 Vt[128 * 40];   // HD x 32 kv (transposed)
  __shared__ alignas(16) bf16 Ps[8 * 16 * 40];// per-wave P staging (A-layout)

  const int tid = threadIdx.x, lane = tid & 31, wave = tid >> 5;
  const int h = blockIdx.y;
  const int hk = h >> 2;                 // G = H/HKV = 4
  const int tok0 = blockIdx.x * 128;     // flat token row (b*N + n)
  const int kvbase = (tok0 >> 11) << 11; // batch base, N = 2048

  // stage Q tile (async)
#pragma unroll
  for (int g = 0; g < 8; ++g) {
    int gid = g * 256 + tid;
    int row = gid >> 4, c = (gid & 15) * 8;
    copy16_to_lds(Qs + row * 136 + c,
                  Q + (size_t)(tok0 + row) * 2048 + h * 128 + c);
  }
  async_wait0();

  const int r = lane & 15, hs = lane >> 4;
  float mrun[8], lrun[8];
#pragma unroll
  for (int i = 0; i < 8; ++i) { mrun[i] = -3.0e38f; lrun[i] = 0.f; }
  v8f oacc[8];
#pragma unroll
  for (int nt = 0; nt < 8; ++nt) oacc[nt] = v8f_zero();

  bf16* Pw = Ps + wave * (16 * 40);
  const float scale = 0.088388347648318447f;  // HD^-0.5

  for (int m0 = 0; m0 < 2048; m0 += 32) {
    __syncthreads();
    // K tile (row-major, async) and V tile (transposed scatter: Vt[d][m])
#pragma unroll
    for (int g = 0; g < 2; ++g) {
      int gid = g * 256 + tid;
      int row = gid >> 4, c = (gid & 15) * 8;
      copy16_to_lds(Ks + row * 136 + c,
                    Kb + (size_t)(kvbase + m0 + row) * 512 + hk * 128 + c);
    }
#pragma unroll
    for (int g = 0; g < 2; ++g) {
      int gid = g * 256 + tid;
      int row = gid >> 4, d0 = (gid & 15) * 8;
      union { v8bf v; bf16 e[8]; } val;
      val.v = *(const v8bf*)(Vb + (size_t)(kvbase + m0 + row) * 512 + hk * 128 + d0);
#pragma unroll
      for (int e = 0; e < 8; ++e) Vt[(d0 + e) * 40 + row] = val.e[e];
    }
    async_wait0();
    __syncthreads();

    // S (16 x 32) = Q_wave x K^T, contracted over HD=128
    v8f s[2];
    s[0] = v8f_zero();
    s[1] = v8f_zero();
#pragma unroll
    for (int ks = 0; ks < 128; ks += 32) {
      v16bf aF = load_frag_a(Qs, 136, wave * 16, ks, lane);
#pragma unroll
      for (int nt = 0; nt < 2; ++nt) {
        v16bf bF = load_frag_b(Ks, 136, nt * 16, ks, lane);
        s[nt] = wmma_bf16(aF, bF, s[nt]);
      }
    }

    // online softmax; row = wave*16 + hs*8 + i, 32 cols across 16 lanes x 2 regs
#pragma unroll
    for (int i = 0; i < 8; ++i) {
      float mx = -3.0e38f;
#pragma unroll
      for (int nt = 0; nt < 2; ++nt) {
        s[nt][i] *= scale;
        mx = fmaxf(mx, s[nt][i]);
      }
      mx = fmaxf(mx, __shfl_xor(mx, 1, 32));
      mx = fmaxf(mx, __shfl_xor(mx, 2, 32));
      mx = fmaxf(mx, __shfl_xor(mx, 4, 32));
      mx = fmaxf(mx, __shfl_xor(mx, 8, 32));
      float mnew = fmaxf(mrun[i], mx);
      float fs = __expf(mrun[i] - mnew);
      mrun[i] = mnew;
      float rsum = 0.f;
#pragma unroll
      for (int nt = 0; nt < 2; ++nt) {
        float p = __expf(s[nt][i] - mnew);
        s[nt][i] = p;
        rsum += p;
      }
      rsum += __shfl_xor(rsum, 1, 32);
      rsum += __shfl_xor(rsum, 2, 32);
      rsum += __shfl_xor(rsum, 4, 32);
      rsum += __shfl_xor(rsum, 8, 32);
      lrun[i] = lrun[i] * fs + rsum;
#pragma unroll
      for (int nt = 0; nt < 8; ++nt) oacc[nt][i] *= fs;
      // C-layout -> A-operand layout via per-wave LDS staging
#pragma unroll
      for (int nt = 0; nt < 2; ++nt)
        Pw[(hs * 8 + i) * 40 + nt * 16 + r] = f2bf(s[nt][i]);
    }

    // O += P (16x32) x V (32x128)
    {
      v16bf aF = load_frag_a(Pw, 40, 0, 0, lane);
#pragma unroll
      for (int nt = 0; nt < 8; ++nt) {
        v16bf bF = load_frag_b(Vt, 40, nt * 16, 0, lane);
        oacc[nt] = wmma_bf16(aF, bF, oacc[nt]);
      }
    }
  }

#pragma unroll
  for (int nt = 0; nt < 8; ++nt)
#pragma unroll
    for (int i = 0; i < 8; ++i) {
      float inv = 1.f / lrun[i];
      int row = tok0 + wave * 16 + hs * 8 + i;
      int col = h * 128 + nt * 16 + r;
      O[(size_t)row * 2048 + col] = f2bf(oacc[nt][i] * inv);
    }
}

// ---------------------------------------------------------------- launcher
extern "C" void kernel_launch(void* const* d_in, const int* in_sizes, int n_in,
                              void* d_out, int out_size, void* d_ws, size_t ws_size,
                              hipStream_t stream) {
  (void)in_sizes; (void)n_in; (void)out_size; (void)ws_size;
  const float* x  = (const float*)d_in[0];
  const float* Wq = (const float*)d_in[1];
  const float* Wk = (const float*)d_in[2];
  const float* Wv = (const float*)d_in[3];
  const float* Wo = (const float*)d_in[4];
  float* out = (float*)d_out;

  const int Btok = 4096;   // B*N
  const int D = 2048, HHD = 2048, KVD = 512;

  char* w = (char*)d_ws;
  bf16* xb  = (bf16*)(w);                        // 4096x2048   (16 MB)
  bf16* wqt = (bf16*)(w + (16ull << 20));        // 2048x2048   ( 8 MB)
  bf16* wkt = (bf16*)(w + (24ull << 20));        //  512x2048   ( 2 MB)
  bf16* wvt = (bf16*)(w + (26ull << 20));        //  512x2048   ( 2 MB)
  bf16* wot = (bf16*)(w + (28ull << 20));        // 2048x2048   ( 8 MB)
  bf16* qb  = (bf16*)(w + (36ull << 20));        // 4096x2048   (16 MB)
  bf16* kb  = (bf16*)(w + (52ull << 20));        // 4096x512    ( 4 MB)
  bf16* vb  = (bf16*)(w + (56ull << 20));        // 4096x512    ( 4 MB)
  bf16* ob  = (bf16*)(w + (60ull << 20));        // 4096x2048   (16 MB) -> 76 MB

  cvt_f32_bf16<<<(Btok * D / 8 + 255) / 256, 256, 0, stream>>>(x, xb, Btok * D / 8);
  tcvt_f32_bf16<<<dim3(D / 32, HHD / 32), 256, 0, stream>>>(Wq, wqt, D, HHD);
  tcvt_f32_bf16<<<dim3(D / 32, KVD / 32), 256, 0, stream>>>(Wk, wkt, D, KVD);
  tcvt_f32_bf16<<<dim3(D / 32, KVD / 32), 256, 0, stream>>>(Wv, wvt, D, KVD);
  tcvt_f32_bf16<<<dim3(HHD / 32, D / 32), 256, 0, stream>>>(Wo, wot, HHD, D);

  gemm_bt<false><<<dim3(HHD / 128, Btok / 128), 256, 0, stream>>>(xb, wqt, qb, Btok, HHD, D);
  gemm_bt<false><<<dim3(KVD / 128, Btok / 128), 256, 0, stream>>>(xb, wkt, kb, Btok, KVD, D);
  gemm_bt<false><<<dim3(KVD / 128, Btok / 128), 256, 0, stream>>>(xb, wvt, vb, Btok, KVD, D);

  rope_inplace<<<(Btok * 16 * 64) / 256, 256, 0, stream>>>(qb, 16);
  rope_inplace<<<(Btok * 4 * 64) / 256, 256, 0, stream>>>(kb, 4);

  attn_fwd<<<dim3(Btok / 128, 16), 256, 0, stream>>>(qb, kb, vb, ob);

  gemm_bt<true><<<dim3(D / 128, Btok / 128), 256, 0, stream>>>(ob, wot, out, Btok, D, HHD);
}
